// nonlocal_attention_9148280341044
// MI455X (gfx1250) — matrix operand order repaired and checked by reading, not compiled, not run on hardware
//
#include <hip/hip_runtime.h>
#include <math.h>
#include <stdint.h>

#define NB   4
#define TT   8
#define CC   512
#define HEADS 8
#define DD   64
#define HW   196
#define SQ   1568            // T*H*W
#define RTOT (NB*SQ)         // 6272
#define NEGVAL (-4294967295.0f)

typedef _Float16 v16h __attribute__((ext_vector_type(16)));
typedef _Float16 v8h  __attribute__((ext_vector_type(8)));
typedef float    v8f  __attribute__((ext_vector_type(8)));

__device__ __forceinline__ v16h cat8(v8h lo, v8h hi) {
  v16h r;
#pragma unroll
  for (int i = 0; i < 8; ++i) { r[i] = lo[i]; r[8 + i] = hi[i]; }
  return r;
}

__device__ __forceinline__ v8f zero8() {
  v8f z = {0.f, 0.f, 0.f, 0.f, 0.f, 0.f, 0.f, 0.f};
  return z;
}

__device__ __forceinline__ v8f wmma_f16(v16h a, v16h b, v8f c) {
  // (neg_a, A, neg_b, B, c_mod, C, reuse_a, reuse_b)
  return __builtin_amdgcn_wmma_f32_16x16x32_f16(false, a, false, b,
                                                (short)0, c, false, false);
}

// ---------------------------------------------------------------------------
// Kernel 1: pack (N,T,C,H,W) f32 -> row-major (RTOT x C) f16 for q/k/v,
// store f32 q_flat, compute sign(|sum_c|) masks for q and k.
// grid (RTOT, 3), block 128
// ---------------------------------------------------------------------------
__global__ void pack_kernel(const float* __restrict__ q,
                            const float* __restrict__ k,
                            const float* __restrict__ v,
                            _Float16* __restrict__ qh,
                            _Float16* __restrict__ kh,
                            _Float16* __restrict__ vh,
                            float* __restrict__ qflat,
                            float* __restrict__ qmask,
                            float* __restrict__ kmask) {
  __shared__ float red[128];
  int r = blockIdx.x;
  int which = blockIdx.y;
  int tid = threadIdx.x;
  int n = r / SQ, s = r % SQ;
  int t = s / HW, hw = s % HW;
  const float* src = (which == 0) ? q : (which == 1) ? k : v;
  _Float16* dst = (which == 0) ? qh : (which == 1) ? kh : vh;
  size_t base = ((size_t)(n * TT + t) * CC) * HW + hw;
  float lsum = 0.f;
#pragma unroll
  for (int i = 0; i < 4; ++i) {
    int c = tid + i * 128;
    float x = src[base + (size_t)c * HW];
    lsum += x;
    dst[(size_t)r * CC + c] = (_Float16)x;
    if (which == 0) qflat[(size_t)r * CC + c] = x;
  }
  red[tid] = lsum;
  __syncthreads();
  for (int off = 64; off > 0; off >>= 1) {
    if (tid < off) red[tid] += red[tid + off];
    __syncthreads();
  }
  if (tid == 0) {
    float m = (red[0] != 0.0f) ? 1.0f : 0.0f;
    if (which == 0) qmask[r] = m;
    else if (which == 1) kmask[r] = m;
  }
}

// ---------------------------------------------------------------------------
// Kernel 2: weight f32 -> f16 (3 x 512x512)
// ---------------------------------------------------------------------------
__global__ void wcvt_kernel(const float* __restrict__ w0,
                            const float* __restrict__ w1,
                            const float* __restrict__ w2,
                            _Float16* __restrict__ h0,
                            _Float16* __restrict__ h1,
                            _Float16* __restrict__ h2) {
  int i = blockIdx.x * blockDim.x + threadIdx.x;  // 0..786431
  int which = i >> 18;                            // 262144 per weight
  int j = i & 262143;
  const float* w = (which == 0) ? w0 : (which == 1) ? w1 : w2;
  _Float16* h = (which == 0) ? h0 : (which == 1) ? h1 : h2;
  h[j] = (_Float16)w[j];
}

// ---------------------------------------------------------------------------
// Kernel 3: projection GEMM  Y = relu(X * W^T + b), f16 in/out, f32 acc.
// grid (49, 8, 3), block 256 (8 waves); each wave: 16 rows x 64 cols.
// ---------------------------------------------------------------------------
__global__ void __launch_bounds__(256)
proj_kernel(const _Float16* __restrict__ xq, const _Float16* __restrict__ xk,
            const _Float16* __restrict__ xv, const _Float16* __restrict__ wq,
            const _Float16* __restrict__ wk, const _Float16* __restrict__ wv,
            const float* __restrict__ bq, const float* __restrict__ bk,
            const float* __restrict__ bv, _Float16* __restrict__ yq,
            _Float16* __restrict__ yk, _Float16* __restrict__ yv) {
  int which = blockIdx.z;
  const _Float16* X = (which == 0) ? xq : (which == 1) ? xk : xv;
  const _Float16* Wt = (which == 0) ? wq : (which == 1) ? wk : wv;
  const float* bias = (which == 0) ? bq : (which == 1) ? bk : bv;
  _Float16* Y = (which == 0) ? yq : (which == 1) ? yk : yv;

  int tid = threadIdx.x;
  int wave = tid >> 5, lane = tid & 31;
  int rowbase = blockIdx.x * 128 + wave * 16;
  int colbase = blockIdx.y * 64;
  int lrow = lane & 15, lhalf = lane >> 4;
  int klo = lhalf * 8;

  const _Float16* arow = X + (size_t)(rowbase + lrow) * CC;
  v8f acc[4] = {zero8(), zero8(), zero8(), zero8()};
  for (int kk = 0; kk < CC; kk += 32) {
    v16h a = cat8(*(const v8h*)(arow + kk + 2 * klo),
                  *(const v8h*)(arow + kk + 8 + 2 * klo));
#pragma unroll
    for (int nt = 0; nt < 4; ++nt) {
      const _Float16* bp =
          Wt + (size_t)(colbase + nt * 16 + lrow) * CC + kk + lhalf * 16;
      v16h b = cat8(*(const v8h*)bp, *(const v8h*)(bp + 8));
      acc[nt] = wmma_f16(a, b, acc[nt]);
    }
  }
  int mrow0 = lhalf * 8;
#pragma unroll
  for (int nt = 0; nt < 4; ++nt) {
    int col = colbase + nt * 16 + lrow;
    float bi = bias[col];
#pragma unroll
    for (int rr = 0; rr < 8; ++rr) {
      float vv = acc[nt][rr] + bi;
      vv = vv > 0.f ? vv : 0.f;
      Y[(size_t)(rowbase + mrow0 + rr) * CC + col] = (_Float16)vv;
    }
  }
}

// ---------------------------------------------------------------------------
// Kernel 4: flash attention per (head,batch) b = h*N + n.
// grid (13, 32), block 256 (8 waves, each owns 16 query rows).
// K staged row-major in LDS via GLOBAL_LOAD_ASYNC_TO_LDS_B128 (ASYNCcnt);
// V staged transposed; P transposed through per-wave LDS (DScnt).
// ---------------------------------------------------------------------------
__global__ void __launch_bounds__(256)
attn_kernel(const _Float16* __restrict__ Qh, const _Float16* __restrict__ Kh,
            const _Float16* __restrict__ Vh, const float* __restrict__ qmask,
            const float* __restrict__ kmask, float* __restrict__ Oacc) {
  __shared__ __attribute__((aligned(16))) _Float16 Kl[32][72];
  __shared__ __attribute__((aligned(16))) _Float16 Vt[64][40];
  __shared__ __attribute__((aligned(16))) _Float16 Pl[8][16][40];

  int b = blockIdx.y;
  int h = b >> 2, nb = b & 3;
  int tid = threadIdx.x;
  int wave = tid >> 5, lane = tid & 31;
  int lrow = lane & 15, lhalf = lane >> 4;
  int klo = lhalf * 8;
  int mbase = lhalf * 8;
  int qbase = blockIdx.x * 128 + wave * 16;

  int qa = qbase + lrow;
  if (qa > SQ - 1) qa = SQ - 1;  // clamp for A loads; stores guarded later
  const _Float16* Qp = Qh + ((size_t)(nb * SQ + qa)) * CC + h * DD;
  v16h a0 = cat8(*(const v8h*)(Qp + 2 * klo), *(const v8h*)(Qp + 8 + 2 * klo));
  v16h a1 = cat8(*(const v8h*)(Qp + 32 + 2 * klo), *(const v8h*)(Qp + 40 + 2 * klo));

  float mr[8], lr[8];
#pragma unroll
  for (int r = 0; r < 8; ++r) { mr[r] = -3.0e38f; lr[r] = 0.f; }
  v8f o0 = zero8(), o1 = zero8(), o2 = zero8(), o3 = zero8();

  int skey = tid >> 3;        // staging: 32 keys x 8 chunks of 8 halfs
  int sd0 = (tid & 7) * 8;
  unsigned kl_off = (unsigned)(uintptr_t)&Kl[skey][sd0];

  for (int kb = 0; kb < SQ; kb += 32) {
    __syncthreads();
    {
      // K tile: async DMA straight into LDS (no VGPR round-trip)
      const _Float16* kp =
          Kh + ((size_t)(nb * SQ + kb + skey)) * CC + h * DD + sd0;
      asm volatile("global_load_async_to_lds_b128 %0, %1, off"
                   :: "v"(kl_off), "v"((unsigned long long)(uintptr_t)kp)
                   : "memory");
      // V tile: load + transpose into LDS
      const _Float16* vp =
          Vh + ((size_t)(nb * SQ + kb + skey)) * CC + h * DD + sd0;
      v8h vv = *(const v8h*)vp;
#pragma unroll
      for (int j = 0; j < 8; ++j) Vt[sd0 + j][skey] = vv[j];
      // prefetch next key block (global_prefetch_b8)
      if (kb + 32 < SQ) {
        __builtin_prefetch(kp + 32 * CC, 0, 1);
        __builtin_prefetch(vp + 32 * CC, 0, 1);
      }
      asm volatile("s_wait_asynccnt 0x0" ::: "memory");
    }
    __syncthreads();

    // S = Q K^T for two 16-key tiles
    float s0[8], s1[8];
#pragma unroll
    for (int t = 0; t < 2; ++t) {
      v8f sacc = zero8();
      const _Float16* bp = &Kl[t * 16 + lrow][lhalf * 16];
      sacc = wmma_f16(a0, cat8(*(const v8h*)bp, *(const v8h*)(bp + 8)), sacc);
      const _Float16* bp2 = &Kl[t * 16 + lrow][32 + lhalf * 16];
      sacc = wmma_f16(a1, cat8(*(const v8h*)bp2, *(const v8h*)(bp2 + 8)), sacc);
      float km = kmask[(size_t)nb * SQ + kb + t * 16 + lrow];
      float* sd = (t == 0) ? s0 : s1;
#pragma unroll
      for (int r = 0; r < 8; ++r)
        sd[r] = (km == 0.f) ? NEGVAL : sacc[r] * 0.125f;  // 1/sqrt(64)
    }

    // online softmax (row = 16-lane half-wave reduction)
    float alpha[8];
#pragma unroll
    for (int r = 0; r < 8; ++r) {
      float mx = fmaxf(s0[r], s1[r]);
      for (int off = 1; off < 16; off <<= 1)
        mx = fmaxf(mx, __shfl_xor(mx, off, 32));
      float mn = fmaxf(mr[r], mx);
      alpha[r] = __expf(mr[r] - mn);
      float p0 = __expf(s0[r] - mn);
      float p1 = __expf(s1[r] - mn);
      float rs = p0 + p1;
      for (int off = 1; off < 16; off <<= 1) rs += __shfl_xor(rs, off, 32);
      lr[r] = lr[r] * alpha[r] + rs;
      mr[r] = mn;
      s0[r] = p0;
      s1[r] = p1;
    }
#pragma unroll
    for (int r = 0; r < 8; ++r) {
      o0[r] *= alpha[r]; o1[r] *= alpha[r];
      o2[r] *= alpha[r]; o3[r] *= alpha[r];
      Pl[wave][mbase + r][lrow] = (_Float16)s0[r];
      Pl[wave][mbase + r][16 + lrow] = (_Float16)s1[r];
    }
    // wave-private D-layout -> A-layout transpose through LDS: drain DS ops
    asm volatile("s_wait_dscnt 0" ::: "memory");

    const _Float16* pa = &Pl[wave][lrow][0];
    v16h A = cat8(*(const v8h*)(pa + 2 * klo), *(const v8h*)(pa + 8 + 2 * klo));
    const _Float16* vb0 = &Vt[0 * 16 + lrow][lhalf * 16];
    o0 = wmma_f16(A, cat8(*(const v8h*)vb0, *(const v8h*)(vb0 + 8)), o0);
    const _Float16* vb1 = &Vt[1 * 16 + lrow][lhalf * 16];
    o1 = wmma_f16(A, cat8(*(const v8h*)vb1, *(const v8h*)(vb1 + 8)), o1);
    const _Float16* vb2 = &Vt[2 * 16 + lrow][lhalf * 16];
    o2 = wmma_f16(A, cat8(*(const v8h*)vb2, *(const v8h*)(vb2 + 8)), o2);
    const _Float16* vb3 = &Vt[3 * 16 + lrow][lhalf * 16];
    o3 = wmma_f16(A, cat8(*(const v8h*)vb3, *(const v8h*)(vb3 + 8)), o3);
  }

  // epilogue: scale by query_mask / l and store to merged-head layout
#pragma unroll
  for (int r = 0; r < 8; ++r) {
    int qg = qbase + mbase + r;
    if (qg >= SQ) continue;
    float qm = qmask[(size_t)nb * SQ + qg];
    float sc = (lr[r] > 0.f) ? (qm / lr[r]) : 0.f;
    size_t ob = ((size_t)(nb * SQ + qg)) * CC + h * DD + lrow;
    Oacc[ob + 0] = o0[r] * sc;
    Oacc[ob + 16] = o1[r] * sc;
    Oacc[ob + 32] = o2[r] * sc;
    Oacc[ob + 48] = o3[r] * sc;
  }
}

// ---------------------------------------------------------------------------
// Kernel 5: residual + LayerNorm (unbiased std) + scatter to (N,T,C,H,W)
// grid RTOT, block 128
// ---------------------------------------------------------------------------
__global__ void lnorm_kernel(const float* __restrict__ Oacc,
                             const float* __restrict__ qflat,
                             const float* __restrict__ gamma,
                             const float* __restrict__ beta,
                             float* __restrict__ out) {
  __shared__ float red1[128];
  __shared__ float red2[128];
  __shared__ float smean, sstd;
  int r = blockIdx.x, tid = threadIdx.x;
  int n = r / SQ, s = r % SQ;
  int t = s / HW, hw = s % HW;
  float vloc[4];
  float sum = 0.f, sq = 0.f;
#pragma unroll
  for (int i = 0; i < 4; ++i) {
    int c = tid + i * 128;
    float v = Oacc[(size_t)r * CC + c] + qflat[(size_t)r * CC + c];
    vloc[i] = v;
    sum += v;
    sq += v * v;
  }
  red1[tid] = sum;
  red2[tid] = sq;
  __syncthreads();
  for (int off = 64; off > 0; off >>= 1) {
    if (tid < off) { red1[tid] += red1[tid + off]; red2[tid] += red2[tid + off]; }
    __syncthreads();
  }
  if (tid == 0) {
    float mean = red1[0] / 512.f;
    float var = (red2[0] - 512.f * mean * mean) / 511.f;  // ddof=1
    var = var > 0.f ? var : 0.f;
    smean = mean;
    sstd = sqrtf(var);
  }
  __syncthreads();
  float mean = smean, invd = 1.f / (sstd + 1e-8f);
#pragma unroll
  for (int i = 0; i < 4; ++i) {
    int c = tid + i * 128;
    float y = gamma[c] * (vloc[i] - mean) * invd + beta[c];
    out[(((size_t)(n * TT + t)) * CC + c) * HW + hw] = y;
  }
}

// ---------------------------------------------------------------------------
extern "C" void kernel_launch(void* const* d_in, const int* in_sizes, int n_in,
                              void* d_out, int out_size, void* d_ws,
                              size_t ws_size, hipStream_t stream) {
  (void)in_sizes; (void)n_in; (void)out_size; (void)ws_size;
  const float* q = (const float*)d_in[0];
  const float* k = (const float*)d_in[1];
  const float* v = (const float*)d_in[2];
  const float* Wq = (const float*)d_in[3];
  const float* bq = (const float*)d_in[4];
  const float* Wk = (const float*)d_in[5];
  const float* bk = (const float*)d_in[6];
  const float* Wv = (const float*)d_in[7];
  const float* bv = (const float*)d_in[8];
  const float* gamma = (const float*)d_in[9];
  const float* beta = (const float*)d_in[10];
  float* out = (float*)d_out;

  char* p = (char*)d_ws;
  auto take = [&](size_t bytes) -> char* {
    char* r = p;
    p += (bytes + 255) & ~(size_t)255;
    return r;
  };
  const size_t RC_H = (size_t)RTOT * CC * sizeof(_Float16);
  const size_t RC_F = (size_t)RTOT * CC * sizeof(float);
  _Float16* xq = (_Float16*)take(RC_H);
  _Float16* xk = (_Float16*)take(RC_H);
  _Float16* xv = (_Float16*)take(RC_H);
  _Float16* whq = (_Float16*)take((size_t)CC * CC * 2);
  _Float16* whk = (_Float16*)take((size_t)CC * CC * 2);
  _Float16* whv = (_Float16*)take((size_t)CC * CC * 2);
  _Float16* qh = (_Float16*)take(RC_H);
  _Float16* kh = (_Float16*)take(RC_H);
  _Float16* vh = (_Float16*)take(RC_H);
  float* qflat = (float*)take(RC_F);
  float* oacc = (float*)take(RC_F);
  float* qm = (float*)take((size_t)RTOT * 4);
  float* km = (float*)take((size_t)RTOT * 4);

  pack_kernel<<<dim3(RTOT, 3), 128, 0, stream>>>(q, k, v, xq, xk, xv, qflat,
                                                 qm, km);
  wcvt_kernel<<<3072, 256, 0, stream>>>(Wq, Wk, Wv, whq, whk, whv);
  proj_kernel<<<dim3(49, 8, 3), 256, 0, stream>>>(xq, xk, xv, whq, whk, whv,
                                                  bq, bk, bv, qh, kh, vh);
  attn_kernel<<<dim3(13, 32), 256, 0, stream>>>(qh, kh, vh, qm, km, oacc);
  lnorm_kernel<<<RTOT, 128, 0, stream>>>(oacc, qflat, gamma, beta, out);
}
